// CRF_59700045414997
// MI455X (gfx1250) — matrix operand order repaired and checked by reading, compiled, not run on hardware
//
#include <hip/hip_runtime.h>
#include <math.h>

typedef __attribute__((ext_vector_type(16))) _Float16 v16h;
typedef __attribute__((ext_vector_type(8)))  float    v8f;

#define T_TAGS 52              // real tag count (incl. START/END)
#define T_PAD  64              // padded to 4 x 16 WMMA tiles
#define ROWS   16              // batch rows per block (one M tile)
#define NEG_INF (-__builtin_inff())
#define LOG2E   1.4426950408889634f
#define LN2     0.6931471805599453f

// raw-hardware transcendentals: v_exp_f32 computes 2^x, v_log_f32 computes log2
__device__ __forceinline__ float fast_exp2(float x) { return __builtin_amdgcn_exp2f(x); }
__device__ __forceinline__ float fast_log2(float x) { return __builtin_amdgcn_logf(x); }

// ---------------------------------------------------------------------------
// CRF forward partition via exp-matmul-log reformulation:
//   part_new = log( exp(part - m) @ exp(trans) ) + m + feats_t
// Inner product runs on v_wmma_f32_16x16x32_f16; one block owns a 16-row
// batch tile, 4 waves split the 64 padded output columns (1 N-tile each).
// Padding columns carry -inf through the recurrence (log(0) = -inf from the
// zero B columns), so the inner loop needs no validity selects at all.
// ---------------------------------------------------------------------------
__global__ __launch_bounds__(128)
void crf_forward_kernel(const float* __restrict__ feats,
                        const int*   __restrict__ mask,
                        const float* __restrict__ trans,
                        float* __restrict__ out,
                        int L)
{
    __shared__ float s_part[ROWS][T_PAD + 4];   // stride 68 floats -> conflict-free
    __shared__ float s_m[ROWS];                 // per-row max (scale) for this step
    __shared__ int   s_mask[ROWS];              // per-row mask at time t

    const int tid = threadIdx.x;
    const int w   = tid >> 5;        // wave id 0..3 (owns cols [16w, 16w+16))
    const int l   = tid & 31;        // lane in wave
    const int r   = l & 15;          // A-phase: row index (M = lane % 16)
    const int h   = l >> 4;          // lane half selects K/M sub-groups
    const int n0  = w * 16;          // this wave's N-tile base column
    const int b0  = blockIdx.x * ROWS;

    const size_t featRowStride = (size_t)L * T_TAGS;

    // ---- constant B fragments: exp(trans) in 32x16 f16 B lane layout ------
    // chunk c covers K = [32c, 32c+32); lane: N = l%16, K = 32c + 16h + j
    v16h bfrag[2];
    #pragma unroll
    for (int c = 0; c < 2; ++c) {
        #pragma unroll
        for (int j = 0; j < 16; ++j) {
            int K   = c * 32 + 16 * h + j;
            int col = n0 + r;
            float e = 0.0f;
            if (K < T_TAGS && col < T_TAGS)
                e = fast_exp2(trans[K * T_TAGS + col] * LOG2E);
            bfrag[c][j] = (_Float16)e;
        }
    }

    // ---- initial partition: feats[:,0,:] + trans[START,:] ------------------
    for (int idx = tid; idx < ROWS * T_PAD; idx += blockDim.x) {
        int rr = idx >> 6;
        int j  = idx & 63;
        float v = NEG_INF;
        if (j < T_TAGS)
            v = feats[(size_t)(b0 + rr) * featRowStride + j] +
                trans[(T_TAGS - 2) * T_TAGS + j];     // START row
        s_part[rr][j] = v;
    }
    __syncthreads();

    // per-lane constants for the writeback (C/D) phase
    const int colD     = n0 + r;                               // output column
    const int colClamp = (colD < T_TAGS) ? colD : (T_TAGS - 1); // safe feats idx

    // full-coverage prefetch mapping: the (t+1) feats tile is 16 rows x 208 B
    // = 32 cachelines; thread -> (row = tid&15, line = (tid>>4)&1)
    const size_t pfBase = (size_t)(b0 + (tid & 15)) * featRowStride +
                          (size_t)(((tid >> 4) & 1) * 32);

    // ---- sequential scan over time ----------------------------------------
    for (int t = 1; t < L; ++t) {
        // issue this step's feats loads up front: latency overlaps the whole
        // A-phase + WMMA, consumed only in the writeback
        float ffrag[8];
        #pragma unroll
        for (int v = 0; v < 8; ++v)
            ffrag[v] = feats[(size_t)(b0 + 8 * h + v) * featRowStride +
                             (size_t)t * T_TAGS + colClamp];
        // speculative prefetch of the NEXT step's full tile (32 lines)
        __builtin_prefetch(&feats[pfBase + (size_t)(t + 1) * T_TAGS], 0, 0);

        // A phase: lane gathers its 32 K-slots of row r (16-bit A striping)
        float vals[32];
        float mrow = NEG_INF;
        #pragma unroll
        for (int c = 0; c < 2; ++c) {
            #pragma unroll
            for (int j = 0; j < 16; ++j) {
                int K = c * 32 + 16 * h + j;
                float v = s_part[r][K];
                vals[c * 16 + j] = v;
                mrow = fmaxf(mrow, v);
            }
        }
        // lane pair (l, l^16) together saw the whole row -> full row max
        mrow = fmaxf(mrow, __shfl_xor(mrow, 16, 32));
        const float mrow2 = mrow * LOG2E;   // row max in log2 units

        // exp(x - m) = exp2(x*log2e - m*log2e): one fma + one v_exp_f32 each
        v16h a0, a1;
        #pragma unroll
        for (int j = 0; j < 16; ++j) {
            a0[j] = (_Float16)fast_exp2(fmaf(vals[j],      LOG2E, -mrow2));
            a1[j] = (_Float16)fast_exp2(fmaf(vals[16 + j], LOG2E, -mrow2));
        }

        if (w == 0 && h == 0) {
            s_m[r]    = mrow;                                   // row scale
            s_mask[r] = mask[(size_t)(b0 + r) * L + t];         // mask at t
        }
        __syncthreads();   // all A reads of s_part done; s_m/s_mask visible

        // GEMM: acc(16x16) = A(16x64) x B(64x16), K split into two WMMAs
        v8f acc = {};
        acc = __builtin_amdgcn_wmma_f32_16x16x32_f16(false, a0, false, bfrag[0],
                                                     (short)0, acc, false, false);
        acc = __builtin_amdgcn_wmma_f32_16x16x32_f16(false, a1, false, bfrag[1],
                                                     (short)0, acc, false, false);

        // contiguous gathers -> ds_load_b128 pairs
        float mrows[8];
        int   msks[8];
        #pragma unroll
        for (int v = 0; v < 8; ++v) {
            mrows[v] = s_m[8 * h + v];
            msks[v]  = s_mask[8 * h + v];
        }

        // writeback: C/D layout -> lane owns col = n0 + l%16, rows v + 8h
        // fully straight-line: log(0) = -inf handles padding columns
        #pragma unroll
        for (int v = 0; v < 8; ++v) {
            int row = 8 * h + v;
            float oldv = s_part[row][colD];
            float newv = fast_log2(acc[v]) * LN2 + mrows[v] + ffrag[v];
            s_part[row][colD] = (msks[v] > 0) ? newv : oldv;
        }
        __syncthreads();
    }

    // ---- final: sum_b logsumexp_i( part[b,i] + trans[i, END] ) -------------
    if (w == 0) {
        float mx = NEG_INF;
        float vv[32];
        #pragma unroll
        for (int k = 0; k < 32; ++k) {
            int i  = 32 * h + k;
            int ic = (i < T_TAGS) ? i : (T_TAGS - 1);
            float v = s_part[r][ic] + trans[ic * T_TAGS + (T_TAGS - 1)];
            vv[k] = (i < T_TAGS) ? v : NEG_INF;
            mx = fmaxf(mx, vv[k]);
        }
        const float mx2u = mx * LOG2E;
        float sm = 0.0f;
        #pragma unroll
        for (int k = 0; k < 32; ++k)
            sm += fast_exp2(fmaf(vv[k], LOG2E, -mx2u));
        // merge the two half-row partials (both halves end up identical)
        float mxo = __shfl_xor(mx, 16, 32);
        float smo = __shfl_xor(sm, 16, 32);
        float mm  = fmaxf(mx, mxo);
        float s   = sm  * fast_exp2((mx  - mm) * LOG2E)
                  + smo * fast_exp2((mxo - mm) * LOG2E);
        float res = fast_log2(s) * LN2 + mm;   // per-row result, row = l%16

        // tree-reduce rows 0..15 into lane 0 (upper-half duplicates never
        // reach lane 0's dependency tree for offsets 8,4,2,1)
        #pragma unroll
        for (int off = 8; off >= 1; off >>= 1)
            res += __shfl_down(res, off, 32);
        if (l == 0) atomicAdd(out, res);
    }
}

__global__ void crf_zero_kernel(float* out) {
    if (threadIdx.x == 0) out[0] = 0.0f;
}

extern "C" void kernel_launch(void* const* d_in, const int* in_sizes, int n_in,
                              void* d_out, int out_size, void* d_ws, size_t ws_size,
                              hipStream_t stream) {
    const float* feats = (const float*)d_in[0];   // (B, L, T) f32
    const int*   mask  = (const int*)d_in[1];     // (B, L)    i32
    const float* trans = (const float*)d_in[2];   // (T, T)    f32
    float*       out   = (float*)d_out;           // scalar

    const int L  = 512;                           // per reference setup
    const int B  = in_sizes[1] / L;               // mask is B*L elements

    crf_zero_kernel<<<1, 1, 0, stream>>>(out);
    crf_forward_kernel<<<B / ROWS, 128, 0, stream>>>(feats, mask, trans, out, L);
}
